// TDSM_token_77996606095432
// MI455X (gfx1250) — matrix operand order repaired and checked by reading, not compiled, not run on hardware
//
#include <hip/hip_runtime.h>
#include <hip/hip_bf16.h>

// ---------------------------------------------------------------------------
// Types / WMMA helpers (CDNA5 / gfx1250, wave32)
// ---------------------------------------------------------------------------
typedef __bf16 bf16_t;
typedef __attribute__((ext_vector_type(16))) __bf16 bf16x16;
typedef __attribute__((ext_vector_type(8)))  float  f32x8;
typedef __attribute__((__vector_size__(4 * sizeof(int)))) int v4i_t;

union FragB16 { bf16x16 v; uint4 q[2]; };

// Async global->LDS path (ASYNCcnt / s_wait_asynccnt), guarded so the build
// can never regress if the toolchain lacks the builtins.
#if defined(__has_builtin)
#if __has_builtin(__builtin_amdgcn_global_load_async_to_lds_b128) && \
    __has_builtin(__builtin_amdgcn_s_wait_asynccnt)
#define HAS_ASYNC_LDS 1
#endif
#endif
#ifndef HAS_ASYNC_LDS
#define HAS_ASYNC_LDS 0
#endif

// A-operand (16x32 bf16): lane holds row M=lane%16.
// Per ISA: lo-lanes K = {0..7, 16..23}, hi-lanes +8. Contiguous 8+8 halves.
__device__ __forceinline__ bf16x16 frag_a_load(const bf16_t* row, int hi) {
    FragB16 f;
    const bf16_t* p = row + (hi ? 8 : 0);
    f.q[0] = *(const uint4*)(p);        // K = kh .. kh+7
    f.q[1] = *(const uint4*)(p + 16);   // K = 16+kh .. 23+kh
    return f.v;
}

// B-operand (32x16 bf16): lane holds column N=lane%16.
// lo-lanes K = {0..7, 16..23}, hi-lanes +8 (same as A).
__device__ __forceinline__ bf16x16 frag_b_load(const bf16_t* col, int hi) {
    FragB16 f;
    const bf16_t* p = col + (hi ? 8 : 0);
    f.q[0] = *(const uint4*)(p);        // K = kh .. kh+7
    f.q[1] = *(const uint4*)(p + 16);   // K = 16+kh .. 23+kh
    return f.v;
}

__device__ __forceinline__ f32x8 wmma_bf16(bf16x16 a, bf16x16 b, f32x8 c) {
    return __builtin_amdgcn_wmma_f32_16x16x32_bf16(false, a, false, b,
                                                   (short)0, c, false, false);
}

// ---------------------------------------------------------------------------
// Problem constants
// ---------------------------------------------------------------------------
#define NB      4        // N
#define LSEQ    4096     // L
#define DMODEL  1152     // D
#define NSEG    4
#define NHEADS  4
#define HD      288      // head dim
#define BWIN    16       // NB * NSEG windows
#define SWIN    1024     // tokens per window
#define MROWS   (BWIN * SWIN)          // 16384
#define N_QKV   (3 * DMODEL)           // 3456
#define TFLAT   (LSEQ * DMODEL)        // 4718592
#define SEGSZ   (SWIN * DMODEL)        // 1179648
#define QSCALE  0.05892556509887896f   // 288^-0.5

// LDS tile geometry for the staged GEMMs: 128x32 bf16 tiles, rows padded to
// 40 elements (80 B = 20 dwords): 16B-aligned rows, bank-conflict-free b128
// fragment reads (stride 20 mod 64 banks covers all 64 banks across 16 lanes).
#define ROWP 40

// ---------------------------------------------------------------------------
// K0: transpose + f32->bf16 convert of a weight matrix W[K][N] -> Wt[N][K]
// ---------------------------------------------------------------------------
__global__ void wt_convert_kernel(const float* __restrict__ W,
                                  bf16_t* __restrict__ Wt, int K, int N) {
    int i = blockIdx.x * 256 + threadIdx.x;
    if (i >= K * N) return;
    int k = i / N, n = i - k * N;
    Wt[(size_t)n * K + k] = (bf16_t)W[i];
}

// ---------------------------------------------------------------------------
// K1: shuffle-gather x into window layout, convert to bf16
// ---------------------------------------------------------------------------
__global__ void gather_kernel(const float* __restrict__ x,
                              const int* __restrict__ idx,
                              bf16_t* __restrict__ xs) {
    int i = blockIdx.x * 256 + threadIdx.x;
    if (i >= MROWS * DMODEL) return;
    int r = i / DMODEL, d = i - r * DMODEL;
    int b = r >> 10, s = r & 1023;
    int nb = b >> 2, sg = b & 3;
    int t = sg * SEGSZ + s * DMODEL + d;
    xs[i] = (bf16_t)x[(size_t)nb * TFLAT + idx[t]];
}

// ---------------------------------------------------------------------------
// Staged GEMM core: 128x128 WG tile, 8 waves (4m x 2n), 32x64 per wave,
// double-buffered 128x32 LDS tiles filled by async global->LDS DMA.
// ---------------------------------------------------------------------------
__device__ __forceinline__ void stage_tile(const bf16_t* __restrict__ gA,
                                           const bf16_t* __restrict__ gB,
                                           bf16_t* lA, bf16_t* lB, int tid) {
#pragma unroll
    for (int i = 0; i < 2; ++i) {
        int c = tid + i * 256;         // 512 16B chunks per 8KB tile
        int row = c >> 2, q = c & 3;
        const bf16_t* ga = gA + (size_t)row * DMODEL + q * 8;
        const bf16_t* gb = gB + (size_t)row * DMODEL + q * 8;
        bf16_t* la = lA + row * ROWP + q * 8;
        bf16_t* lb = lB + row * ROWP + q * 8;
#if HAS_ASYNC_LDS
        __builtin_amdgcn_global_load_async_to_lds_b128((v4i_t*)ga, (v4i_t*)la, 0, 0);
        __builtin_amdgcn_global_load_async_to_lds_b128((v4i_t*)gb, (v4i_t*)lb, 0, 0);
#else
        uint4 va = *(const uint4*)ga;
        uint4 vb = *(const uint4*)gb;
        *(uint4*)la = va;
        *(uint4*)lb = vb;
#endif
    }
}

__device__ __forceinline__ void gemm_core(const bf16_t* __restrict__ A,
                                          const bf16_t* __restrict__ Bt,
                                          bf16_t* lA, bf16_t* lB,   // [2][128*ROWP]
                                          int m_blk, int n_blk,
                                          f32x8 acc[2][4]) {
    const int tid  = threadIdx.x;
    const int lane = tid & 31;
    const int wave = tid >> 5;
    const int mw = wave & 3, nw = wave >> 2;
    const int lr = lane & 15, hi = lane >> 4;

    const bf16_t* gA = A  + (size_t)m_blk * DMODEL;
    const bf16_t* gB = Bt + (size_t)n_blk * DMODEL;

    stage_tile(gA, gB, lA, lB, tid);

    const int nk = DMODEL / 32;  // 36
    for (int kt = 0; kt < nk; ++kt) {
        int cur = kt & 1;
        if (kt + 1 < nk)
            stage_tile(gA + (kt + 1) * 32, gB + (kt + 1) * 32,
                       lA + (cur ^ 1) * (128 * ROWP),
                       lB + (cur ^ 1) * (128 * ROWP), tid);
#if HAS_ASYNC_LDS
        if (kt + 1 < nk) __builtin_amdgcn_s_wait_asynccnt(4);
        else             __builtin_amdgcn_s_wait_asynccnt(0);
#endif
        __syncthreads();

        const bf16_t* pA = lA + cur * (128 * ROWP) + (mw * 32 + lr) * ROWP;
        bf16x16 a0 = frag_a_load(pA, hi);
        bf16x16 a1 = frag_a_load(pA + 16 * ROWP, hi);
        const bf16_t* pB = lB + cur * (128 * ROWP) + (nw * 64 + lr) * ROWP;
#pragma unroll
        for (int j = 0; j < 4; ++j) {
            bf16x16 b = frag_b_load(pB + j * 16 * ROWP, hi);
            acc[0][j] = wmma_bf16(a0, b, acc[0][j]);
            acc[1][j] = wmma_bf16(a1, b, acc[1][j]);
        }
        __syncthreads();
    }
}

// ---------------------------------------------------------------------------
// K2: QKV GEMM.  C[16384,3456] = xs @ qkv_w + b.
// Epilogue scatters bf16 q (pre-scaled), k, and transposed v.
// ---------------------------------------------------------------------------
__global__ __launch_bounds__(256)
void qkv_gemm_kernel(const bf16_t* __restrict__ A,    // [16384 x 1152]
                     const bf16_t* __restrict__ Bt,   // [3456 x 1152] (transposed)
                     const float*  __restrict__ bias, // [3456]
                     bf16_t* __restrict__ qb,         // [16,4,1024,288]
                     bf16_t* __restrict__ kb,         // [16,4,1024,288]
                     bf16_t* __restrict__ vtb) {      // [16,4,288,1024]
    __shared__ __align__(16) bf16_t lA[2][128 * ROWP];
    __shared__ __align__(16) bf16_t lB[2][128 * ROWP];

    const int lane = threadIdx.x & 31;
    const int wave = threadIdx.x >> 5;
    const int mw = wave & 3, nw = wave >> 2;
    const int m0 = blockIdx.x * 128 + mw * 32;
    const int n0 = blockIdx.y * 128 + nw * 64;
    const int lr = lane & 15, hi = lane >> 4;

    f32x8 acc[2][4];
#pragma unroll
    for (int i = 0; i < 2; ++i)
#pragma unroll
        for (int j = 0; j < 4; ++j) acc[i][j] = (f32x8){0,0,0,0,0,0,0,0};

    gemm_core(A, Bt, &lA[0][0], &lB[0][0], blockIdx.x * 128, blockIdx.y * 128, acc);

#pragma unroll
    for (int i = 0; i < 2; ++i) {
#pragma unroll
        for (int j = 0; j < 4; ++j) {
            int col = n0 + j * 16 + lr;
            int m3  = col / DMODEL;
            int rem = col - m3 * DMODEL;
            int h   = rem / HD;
            int d   = rem - h * HD;
            float bi = bias[col];
#pragma unroll
            for (int r = 0; r < 8; ++r) {
                int m = m0 + i * 16 + r + hi * 8;
                int b = m >> 10, s = m & 1023;
                float val = acc[i][j][r] + bi;
                size_t bh = (size_t)(b * NHEADS + h);
                if (m3 == 0)
                    qb[(bh * SWIN + s) * HD + d] = (bf16_t)(val * QSCALE);
                else if (m3 == 1)
                    kb[(bh * SWIN + s) * HD + d] = (bf16_t)val;
                else
                    vtb[(bh * HD + d) * SWIN + s] = (bf16_t)val;
            }
        }
    }
}

// ---------------------------------------------------------------------------
// K3: flash attention.  block = 4 waves, each wave owns a 16-row Q tile.
// grid = (S/64, B*H).  Streams 32 keys per iteration:
//   18 QK^T WMMAs -> online softmax -> P via LDS relayout -> 18 PV WMMAs.
// ---------------------------------------------------------------------------
__global__ __launch_bounds__(128)
void attn_kernel(const bf16_t* __restrict__ qb,
                 const bf16_t* __restrict__ kb,
                 const bf16_t* __restrict__ vtb,
                 bf16_t* __restrict__ ao) {          // [16384 x 1152]
    const int lane = threadIdx.x & 31;
    const int wave = threadIdx.x >> 5;
    const int lr = lane & 15, hi = lane >> 4;
    const int bh = blockIdx.y;
    const int b  = bh >> 2, h = bh & 3;
    const int q0 = blockIdx.x * 64 + wave * 16;

    __shared__ __align__(16) bf16_t pstage[4][16 * 32];
    bf16_t* ps = pstage[wave];

    const bf16_t* qbase = qb  + (size_t)bh * SWIN * HD;
    const bf16_t* kbase = kb  + (size_t)bh * SWIN * HD;
    const bf16_t* vbase = vtb + (size_t)bh * HD * SWIN;

    bf16x16 qf[9];
    const bf16_t* qrow = qbase + (size_t)(q0 + lr) * HD;
#pragma unroll
    for (int c = 0; c < 9; ++c) qf[c] = frag_a_load(qrow + c * 32, hi);

    f32x8 oacc[18];
#pragma unroll
    for (int n = 0; n < 18; ++n) oacc[n] = (f32x8){0,0,0,0,0,0,0,0};
    float mrun[8], lrun[8];
#pragma unroll
    for (int r = 0; r < 8; ++r) { mrun[r] = -1e30f; lrun[r] = 0.f; }

    for (int kt = 0; kt < SWIN / 32; ++kt) {
        f32x8 s0 = (f32x8){0,0,0,0,0,0,0,0};
        f32x8 s1 = (f32x8){0,0,0,0,0,0,0,0};
        const bf16_t* k0row = kbase + (size_t)(kt * 32 + lr) * HD;
        const bf16_t* k1row = k0row + (size_t)16 * HD;
#pragma unroll
        for (int c = 0; c < 9; ++c) {
            s0 = wmma_bf16(qf[c], frag_b_load(k0row + c * 32, hi), s0);
            s1 = wmma_bf16(qf[c], frag_b_load(k1row + c * 32, hi), s1);
        }
        // online softmax (C layout: VGPR r -> row r + 8*hi, 16 cols across
        // the 16 lanes of each half-wave)
#pragma unroll
        for (int r = 0; r < 8; ++r) {
            float mx = fmaxf(s0[r], s1[r]);
#pragma unroll
            for (int off = 1; off < 16; off <<= 1)
                mx = fmaxf(mx, __shfl_xor(mx, off, 32));
            float mnew  = fmaxf(mrun[r], mx);
            float alpha = __expf(mrun[r] - mnew);
            float p0 = __expf(s0[r] - mnew);
            float p1 = __expf(s1[r] - mnew);
            s0[r] = p0; s1[r] = p1;
            float rs = p0 + p1;
#pragma unroll
            for (int off = 1; off < 16; off <<= 1)
                rs += __shfl_xor(rs, off, 32);
            lrun[r] = lrun[r] * alpha + rs;
            mrun[r] = mnew;
#pragma unroll
            for (int n = 0; n < 18; ++n) oacc[n][r] *= alpha;
        }
        // relayout P: C layout -> LDS [row][col] -> A fragment (16x32 bf16)
#pragma unroll
        for (int r = 0; r < 8; ++r) {
            int row = r + hi * 8;
            ps[row * 32 + lr]      = (bf16_t)s0[r];
            ps[row * 32 + 16 + lr] = (bf16_t)s1[r];
        }
        asm volatile("s_wait_dscnt 0" ::: "memory");
        FragB16 pa;
        const bf16_t* pr = ps + lr * 32 + (hi ? 8 : 0);
        pa.q[0] = *(const uint4*)(pr);
        pa.q[1] = *(const uint4*)(pr + 16);
        // PV: B operand straight from transposed V (contiguous keys per lane)
        const bf16_t* vk = vbase + kt * 32;
#pragma unroll
        for (int n = 0; n < 18; ++n) {
            const bf16_t* vrow = vk + (size_t)(n * 16 + lr) * SWIN;
            oacc[n] = wmma_bf16(pa.v, frag_b_load(vrow, hi), oacc[n]);
        }
    }

    float inv[8];
#pragma unroll
    for (int r = 0; r < 8; ++r) inv[r] = 1.f / lrun[r];
#pragma unroll
    for (int n = 0; n < 18; ++n) {
        int col = h * HD + n * 16 + lr;
#pragma unroll
        for (int r = 0; r < 8; ++r) {
            int m = q0 + r + hi * 8;
            ao[((size_t)(b * SWIN + m)) * DMODEL + col] =
                (bf16_t)(oacc[n][r] * inv[r]);
        }
    }
}

// ---------------------------------------------------------------------------
// K4: proj GEMM with fused inverse-permutation scatter + expertCoef.
// ---------------------------------------------------------------------------
__global__ __launch_bounds__(256)
void proj_gemm_kernel(const bf16_t* __restrict__ A,    // [16384 x 1152]
                      const bf16_t* __restrict__ Bt,   // [1152 x 1152] (transposed)
                      const float*  __restrict__ bias, // [1152]
                      const int*    __restrict__ idx,  // [T]
                      const int*    __restrict__ coefp,
                      float* __restrict__ out) {
    __shared__ __align__(16) bf16_t lA[2][128 * ROWP];
    __shared__ __align__(16) bf16_t lB[2][128 * ROWP];

    const int lane = threadIdx.x & 31;
    const int wave = threadIdx.x >> 5;
    const int mw = wave & 3, nw = wave >> 2;
    const int m0 = blockIdx.x * 128 + mw * 32;
    const int n0 = blockIdx.y * 128 + nw * 64;
    const int lr = lane & 15, hi = lane >> 4;

    f32x8 acc[2][4];
#pragma unroll
    for (int i = 0; i < 2; ++i)
#pragma unroll
        for (int j = 0; j < 4; ++j) acc[i][j] = (f32x8){0,0,0,0,0,0,0,0};

    gemm_core(A, Bt, &lA[0][0], &lB[0][0], blockIdx.x * 128, blockIdx.y * 128, acc);

    const float coef = (float)coefp[0];
#pragma unroll
    for (int i = 0; i < 2; ++i) {
#pragma unroll
        for (int j = 0; j < 4; ++j) {
            int col = n0 + j * 16 + lr;
            float bi = bias[col];
#pragma unroll
            for (int r = 0; r < 8; ++r) {
                int m = m0 + i * 16 + r + hi * 8;
                int b = m >> 10, s = m & 1023;
                int nb = b >> 2, sg = b & 3;
                int t = sg * SEGSZ + s * DMODEL + col;
                out[(size_t)nb * TFLAT + idx[t]] = (acc[i][j][r] + bi) * coef;
            }
        }
    }
}

// ---------------------------------------------------------------------------
// Launch
// ---------------------------------------------------------------------------
extern "C" void kernel_launch(void* const* d_in, const int* in_sizes, int n_in,
                              void* d_out, int out_size, void* d_ws, size_t ws_size,
                              hipStream_t stream) {
    const float* x      = (const float*)d_in[0];
    const float* qkv_w  = (const float*)d_in[1];
    const float* qkv_b  = (const float*)d_in[2];
    const float* proj_w = (const float*)d_in[3];
    const float* proj_b = (const float*)d_in[4];
    const int*   sidx   = (const int*)  d_in[5];
    const int*   coefp  = (const int*)  d_in[6];
    float* out = (float*)d_out;

    // workspace layout (bytes, all 256-aligned)
    char* ws = (char*)d_ws;
    bf16_t* wqT = (bf16_t*)(ws + 0);          //  3456*1152*2 =  7962624
    bf16_t* wpT = (bf16_t*)(ws + 7962624);    //  1152*1152*2 =  2654208
    bf16_t* xs  = (bf16_t*)(ws + 10616832);   // 16384*1152*2 = 37748736
    bf16_t* qb  = (bf16_t*)(ws + 48365568);   // 37748736
    bf16_t* kb  = (bf16_t*)(ws + 86114304);   // 37748736
    bf16_t* vtb = (bf16_t*)(ws + 123863040);  // 37748736
    bf16_t* ao  = (bf16_t*)(ws + 161611776);  // 37748736  -> total 199360512

    // K0: weight transpose + convert
    wt_convert_kernel<<<(DMODEL * N_QKV + 255) / 256, 256, 0, stream>>>(
        qkv_w, wqT, DMODEL, N_QKV);
    wt_convert_kernel<<<(DMODEL * DMODEL + 255) / 256, 256, 0, stream>>>(
        proj_w, wpT, DMODEL, DMODEL);

    // K1: shuffle gather + convert
    gather_kernel<<<(MROWS * DMODEL + 255) / 256, 256, 0, stream>>>(x, sidx, xs);

    // K2: QKV GEMM (WMMA bf16, async-LDS staged) + scatter epilogue
    qkv_gemm_kernel<<<dim3(MROWS / 128, N_QKV / 128), 256, 0, stream>>>(
        xs, wqT, qkv_b, qb, kb, vtb);

    // K3: flash attention (WMMA bf16)
    attn_kernel<<<dim3(SWIN / 64, BWIN * NHEADS), 128, 0, stream>>>(
        qb, kb, vtb, ao);

    // K4: proj GEMM (WMMA bf16, async-LDS staged) + fused inverse shuffle scatter
    proj_gemm_kernel<<<dim3(MROWS / 128, DMODEL / 128), 256, 0, stream>>>(
        ao, wpT, proj_b, sidx, coefp, out);
}
